// RNN_50148038148722
// MI455X (gfx1250) — matrix operand order repaired and checked by reading, not compiled, not run on hardware
//
#include <hip/hip_runtime.h>

// ---------------- problem constants ----------------
#define L_LAYERS 2
#define B_   128
#define T_   64
#define H_   1024
#define EMB_ 512
#define CNN_ 512
#define V_   10000
#define K2_  2048   // IS_HSS = EMB+CNN+H = H+H

// ---------------- WMMA types ----------------
typedef __attribute__((ext_vector_type(16))) __bf16         v16bf;
typedef __attribute__((ext_vector_type(8)))  float          v8f;
typedef __attribute__((ext_vector_type(8)))  unsigned short us8;

union FragU {
    us8   h[2];   // two 16-byte chunks (8 bf16 each)
    v16bf v;
};

// float -> bf16 (round to nearest even)
__device__ __forceinline__ unsigned short f2bf(float x) {
    unsigned int u = __float_as_uint(x);
    unsigned int r = u + 0x7FFFu + ((u >> 16) & 1u);
    return (unsigned short)(r >> 16);
}

// ----------------------------------------------------------------------
// Core: one wave accumulates a 16 x (NT*16) strip of C = A * Wt^T.
//   A  : row-major [M, lda] bf16 (as ushort)
//   Wt : row-major [N, K]  bf16 (as ushort)  (B-matrix columns contiguous)
// Fragment layouts per CDNA5 ISA 7.12.2 (16-bit A 16x32; B 32x16).
// NT is a compile-time constant: the K-loop body is branch-free.
// ----------------------------------------------------------------------
template <int NT>
__device__ __forceinline__ void gemm_strip(
    const unsigned short* __restrict__ A, int lda,
    const unsigned short* __restrict__ Wt, int K,
    int m0, int n0, int row, int sel, v8f* acc)
{
    const unsigned short* arow =
        A + (size_t)(m0 + row) * (size_t)lda + sel * 8;
    const unsigned short* bp[NT];
#pragma unroll
    for (int nt = 0; nt < NT; ++nt)
        bp[nt] = Wt + (size_t)(n0 + nt * 16 + row) * (size_t)K + sel * 8;

#pragma unroll 2
    for (int kk = 0; kk < K; kk += 32) {
        FragU a;
        a.h[0] = *(const us8*)(arow);        // K = kk + sel*8 .. +7
        a.h[1] = *(const us8*)(arow + 16);   // K = kk + sel*8 + 16 .. +23
        arow += 32;
#pragma unroll
        for (int nt = 0; nt < NT; ++nt) {
            FragU b;
            b.h[0] = *(const us8*)(bp[nt]);      // K = kk + sel*8 .. +7
            b.h[1] = *(const us8*)(bp[nt] + 16); // K = kk + sel*8 + 16 .. +23
            bp[nt] += 32;
            acc[nt] = __builtin_amdgcn_wmma_f32_16x16x32_bf16(
                false, a.v, false, b.v, (short)0, acc[nt], false, false);
        }
    }
}

// ----------------------------------------------------------------------
// One-time prep kernels (run every launch; determinism rule)
// ----------------------------------------------------------------------

// fp32 [K,N] -> bf16 [N,K] (transpose=1, gate weights) or fp32 [N,K] -> bf16
// [N,K] (transpose=0, W_out). Optional column bias folded in ((Wc+bc) quirk).
__global__ void convert_w_kernel(const float* __restrict__ src,
                                 const float* __restrict__ bias,
                                 unsigned short* __restrict__ dst,
                                 int K, int N, int do_transpose, int total)
{
    int idx = blockIdx.x * blockDim.x + threadIdx.x;
    if (idx >= total) return;
    int n = idx / K;
    int k = idx - n * K;
    float v = do_transpose ? src[(size_t)k * N + n] : src[(size_t)n * K + k];
    if (bias) v += bias[n];
    dst[(size_t)n * K + k] = f2bf(v);
}

// x0 = [emb_table[tokens[:,0]], cnn] -> bf16 into first halves of layer-0 A bufs
__global__ void build_x0_kernel(const int* __restrict__ tokens,
                                const float* __restrict__ cnn,
                                const float* __restrict__ emb,
                                unsigned short* __restrict__ xh0,
                                unsigned short* __restrict__ xh2_0)
{
    int idx = blockIdx.x * blockDim.x + threadIdx.x;
    if (idx >= B_ * H_) return;
    int b = idx >> 10;
    int j = idx & 1023;
    float v = (j < EMB_) ? emb[(size_t)tokens[b * T_] * EMB_ + j]
                         : cnn[(size_t)b * CNN_ + (j - EMB_)];
    unsigned short bv = f2bf(v);
    xh0[(size_t)b * K2_ + j]   = bv;
    xh2_0[(size_t)b * K2_ + j] = bv;
}

// h_ws = initial_hidden_state (fp32); bf16 copy into second halves of xh bufs
__global__ void init_h_kernel(const float* __restrict__ h_in,
                              float* __restrict__ h_ws,
                              unsigned short* __restrict__ xh0,
                              unsigned short* __restrict__ xh1)
{
    int idx = blockIdx.x * blockDim.x + threadIdx.x;
    if (idx >= L_LAYERS * B_ * H_) return;
    float v = h_in[idx];
    h_ws[idx] = v;
    int l   = idx / (B_ * H_);
    int rem = idx - l * (B_ * H_);
    int b = rem >> 10;
    int j = rem & 1023;
    unsigned short* x = l ? xh1 : xh0;
    x[(size_t)b * K2_ + H_ + j] = f2bf(v);
}

// ----------------------------------------------------------------------
// GEMM kernels (block = 256 threads = 8 waves; wave w -> rows [16w,16w+16))
// N = 1024 for gate/c kernels -> grid of exactly 16 blocks, no bound checks.
// ----------------------------------------------------------------------

// MODE 0: u = sigmoid(xh@Wu + bu)        -> u_buf (fp32)
// MODE 1: r = sigmoid(xh@Wr + br); r*h   -> bf16 into xh2 second half
template <int MODE>
__global__ void gemm_gate_kernel(const unsigned short* __restrict__ A,
                                 const unsigned short* __restrict__ Wt,
                                 const float* __restrict__ bias,
                                 const float* __restrict__ h_l,
                                 float* __restrict__ u_buf,
                                 unsigned short* __restrict__ rh_dst)
{
    int lane = threadIdx.x & 31;
    int wv   = threadIdx.x >> 5;
    int row  = lane & 15;
    int sel  = lane >> 4;
    int m0   = wv * 16;
    int n0   = blockIdx.x * 64;

    v8f acc[4] = {};
    gemm_strip<4>(A, K2_, Wt, K2_, m0, n0, row, sel, acc);

#pragma unroll
    for (int nt = 0; nt < 4; ++nt) {
#pragma unroll
        for (int r = 0; r < 8; ++r) {
            int m = m0 + sel * 8 + r;
            int n = n0 + nt * 16 + row;
            float pre = acc[nt][r] + bias[n];
            float s   = 1.0f / (1.0f + __expf(-pre));
            if (MODE == 0) {
                u_buf[(size_t)m * H_ + n] = s;
            } else {
                float rh = s * h_l[(size_t)m * H_ + n];
                rh_dst[(size_t)m * K2_ + H_ + n] = f2bf(rh);
            }
        }
    }
}

// c = tanh(xh2 @ (Wc+bc));  h_new = u*h + (1-u)*c.
// LAYER 0: fan bf16(h_new) -> xh1 first half, xh2_1 first half, xh0 second half
// LAYER 1: fan bf16(h_new) -> xh1 second half
template <int LAYER>
__global__ void gemm_c_kernel(const unsigned short* __restrict__ A,
                              const unsigned short* __restrict__ Wt,
                              const float* __restrict__ u_buf,
                              float* __restrict__ h_l,
                              unsigned short* __restrict__ t0,
                              unsigned short* __restrict__ t1,
                              unsigned short* __restrict__ t2)
{
    int lane = threadIdx.x & 31;
    int wv   = threadIdx.x >> 5;
    int row  = lane & 15;
    int sel  = lane >> 4;
    int m0   = wv * 16;
    int n0   = blockIdx.x * 64;

    v8f acc[4] = {};
    gemm_strip<4>(A, K2_, Wt, K2_, m0, n0, row, sel, acc);

#pragma unroll
    for (int nt = 0; nt < 4; ++nt) {
#pragma unroll
        for (int r = 0; r < 8; ++r) {
            int m = m0 + sel * 8 + r;
            int n = n0 + nt * 16 + row;
            float c  = tanhf(acc[nt][r]);
            float u  = u_buf[(size_t)m * H_ + n];
            float hv = h_l[(size_t)m * H_ + n];
            float hn = u * hv + (1.0f - u) * c;
            h_l[(size_t)m * H_ + n] = hn;
            unsigned short hb = f2bf(hn);
            if (LAYER == 0) {
                t0[(size_t)m * K2_ + n] = hb;
                t1[(size_t)m * K2_ + n] = hb;
                t2[(size_t)m * K2_ + n] = hb;
            } else {
                t0[(size_t)m * K2_ + n] = hb;
            }
        }
    }
}

// logits[:, t, :] = h1 @ W_out^T + b_out.
// N = 10000 = 156 full 64-col blocks (NT=4) + one 16-col tail block (NT=1).
template <int NT>
__global__ void gemm_out_kernel(const unsigned short* __restrict__ A,
                                const unsigned short* __restrict__ Wt,
                                const float* __restrict__ b_out,
                                float* __restrict__ out, int ldo, int n_base)
{
    int lane = threadIdx.x & 31;
    int wv   = threadIdx.x >> 5;
    int row  = lane & 15;
    int sel  = lane >> 4;
    int m0   = wv * 16;
    int n0   = n_base + blockIdx.x * 64;

    v8f acc[NT] = {};
    gemm_strip<NT>(A, K2_, Wt, H_, m0, n0, row, sel, acc);

#pragma unroll
    for (int nt = 0; nt < NT; ++nt) {
#pragma unroll
        for (int r = 0; r < 8; ++r) {
            int m = m0 + sel * 8 + r;
            int n = n0 + nt * 16 + row;
            out[(size_t)m * ldo + n] = acc[nt][r] + b_out[n];
        }
    }
}

// ----------------------------------------------------------------------
// Host orchestration
// ----------------------------------------------------------------------
extern "C" void kernel_launch(void* const* d_in, const int* in_sizes, int n_in,
                              void* d_out, int out_size, void* d_ws, size_t ws_size,
                              hipStream_t stream)
{
    (void)in_sizes; (void)n_in; (void)out_size; (void)ws_size;

    const int*   tokens = (const int*)  d_in[0];
    const float* cnn    = (const float*)d_in[1];
    const float* h_init = (const float*)d_in[2];
    const float* Wu     = (const float*)d_in[3];
    const float* bu     = (const float*)d_in[4];
    const float* Wr     = (const float*)d_in[5];
    const float* br     = (const float*)d_in[6];
    const float* Wc     = (const float*)d_in[7];
    const float* bc     = (const float*)d_in[8];
    const float* W_out  = (const float*)d_in[9];
    const float* b_out  = (const float*)d_in[10];
    const float* emb    = (const float*)d_in[11];

    // ---- workspace carve-up (256B aligned regions) ----
    char*  ws  = (char*)d_ws;
    size_t off = 0;
    auto alloc = [&](size_t bytes) -> void* {
        void* p = ws + off;
        off += (bytes + 255) & ~(size_t)255;
        return p;
    };
    const size_t tg = (size_t)K2_ * H_;          // per-layer gate weight elems
    unsigned short* wt_u  = (unsigned short*)alloc(L_LAYERS * tg * 2);
    unsigned short* wt_r  = (unsigned short*)alloc(L_LAYERS * tg * 2);
    unsigned short* wt_c  = (unsigned short*)alloc(L_LAYERS * tg * 2);
    unsigned short* wt_o  = (unsigned short*)alloc((size_t)V_ * H_ * 2);
    unsigned short* xh0   = (unsigned short*)alloc((size_t)B_ * K2_ * 2);
    unsigned short* xh1   = (unsigned short*)alloc((size_t)B_ * K2_ * 2);
    unsigned short* xh2_0 = (unsigned short*)alloc((size_t)B_ * K2_ * 2);
    unsigned short* xh2_1 = (unsigned short*)alloc((size_t)B_ * K2_ * 2);
    float*          h_ws  = (float*)alloc((size_t)L_LAYERS * B_ * H_ * 4);
    float*          u_buf = (float*)alloc((size_t)B_ * H_ * 4);

    unsigned short* xh[2]  = {xh0, xh1};
    unsigned short* xh2[2] = {xh2_0, xh2_1};

    const dim3 blk(256);

    // ---- weight conversion: fp32 [K,N] -> bf16 [N,K] (+ bc folded into Wc) ----
    const int tgi = (int)tg;
    for (int l = 0; l < L_LAYERS; ++l) {
        convert_w_kernel<<<(tgi + 255) / 256, blk, 0, stream>>>(
            Wu + l * tg, nullptr,     wt_u + l * tg, K2_, H_, 1, tgi);
        convert_w_kernel<<<(tgi + 255) / 256, blk, 0, stream>>>(
            Wr + l * tg, nullptr,     wt_r + l * tg, K2_, H_, 1, tgi);
        convert_w_kernel<<<(tgi + 255) / 256, blk, 0, stream>>>(
            Wc + l * tg, bc + l * H_, wt_c + l * tg, K2_, H_, 1, tgi);
    }
    const int to = V_ * H_;
    convert_w_kernel<<<(to + 255) / 256, blk, 0, stream>>>(
        W_out, nullptr, wt_o, H_, V_, 0, to);

    // ---- activations: x0 (constant over t, per the reference quirk) + h init ----
    build_x0_kernel<<<(B_ * H_ + 255) / 256, blk, 0, stream>>>(
        tokens, cnn, emb, xh0, xh2_0);
    init_h_kernel<<<(L_LAYERS * B_ * H_ + 255) / 256, blk, 0, stream>>>(
        h_init, h_ws, xh0, xh1);

    // ---- recurrence ----
    for (int t = 0; t < T_; ++t) {
        for (int l = 0; l < L_LAYERS; ++l) {
            float* hl = h_ws + (size_t)l * B_ * H_;
            // u gate
            gemm_gate_kernel<0><<<dim3(H_ / 64), blk, 0, stream>>>(
                xh[l], wt_u + l * tg, bu + l * H_, hl, u_buf, nullptr);
            // r gate + r*h into xh2 second half
            gemm_gate_kernel<1><<<dim3(H_ / 64), blk, 0, stream>>>(
                xh[l], wt_r + l * tg, br + l * H_, hl, nullptr, xh2[l]);
            // c + h update; fan new h into dependent bf16 A-buffers
            if (l == 0) {
                gemm_c_kernel<0><<<dim3(H_ / 64), blk, 0, stream>>>(
                    xh2[0], wt_c + 0 * tg, u_buf, hl,
                    xh[1], xh2[1], xh[0] + H_);
            } else {
                gemm_c_kernel<1><<<dim3(H_ / 64), blk, 0, stream>>>(
                    xh2[1], wt_c + 1 * tg, u_buf, hl,
                    xh[1] + H_, nullptr, nullptr);
            }
        }
        // output projection for this timestep (A = h1 in xh1 second half)
        float* out_t = (float*)d_out + (size_t)t * V_;
        gemm_out_kernel<4><<<dim3(V_ / 64), blk, 0, stream>>>(
            xh[1] + H_, wt_o, b_out, out_t, T_ * V_, 0);
        gemm_out_kernel<1><<<dim3(1), blk, 0, stream>>>(
            xh[1] + H_, wt_o, b_out, out_t, T_ * V_, (V_ / 64) * 64);
    }

    // ---- h_final appended after logits in d_out ----
    hipMemcpyAsync((float*)d_out + (size_t)B_ * T_ * V_, h_ws,
                   (size_t)L_LAYERS * B_ * H_ * sizeof(float),
                   hipMemcpyDeviceToDevice, stream);
}
